// ContrastiveLoss_6700148982350
// MI455X (gfx1250) — matrix operand order repaired and checked by reading, not compiled, not run on hardware
//
#include <hip/hip_runtime.h>
#include <stdint.h>

// Problem constants (from reference: N=4096, V=2, D=256, T=0.07)
#define NSMP    4096
#define NVIEW   2
#define DDIM    256
#define MROWS   (NSMP * NVIEW)      // 8192 rows in Af/Bf
#define NTILES  (MROWS / 16)        // 512 16-wide tiles
#define NCHUNK  4                   // column-tile chunks (occupancy)
#define CTPER   (NTILES / NCHUNK)   // 128 col tiles per chunk
#define LDSPITCH 264                // 256 + 8 ushorts pad -> conflict-free b128 reads
#define INV_T   (1.0f / 0.07f)

typedef __attribute__((ext_vector_type(16))) __bf16          v16bf;
typedef __attribute__((ext_vector_type(8)))  float           v8f;
typedef __attribute__((ext_vector_type(8)))  unsigned short  ushort8;

__device__ __forceinline__ unsigned short f2bf(float f) {
  unsigned int u = __float_as_uint(f);
  u += 0x7FFFu + ((u >> 16) & 1u);          // round-to-nearest-even
  return (unsigned short)(u >> 16);
}
__device__ __forceinline__ float bf2f(unsigned short h) {
  return __uint_as_float(((unsigned int)h) << 16);
}

// ---------------------------------------------------------------------------
// Pass 1: L2-normalize each view-major row and split fp32 -> bf16 hi + lo.
// One wave per row; lane handles 8 contiguous elements.
// ---------------------------------------------------------------------------
__global__ void __launch_bounds__(32)
norm_split_kernel(const float* __restrict__ A, const float* __restrict__ B,
                  unsigned short* __restrict__ AH, unsigned short* __restrict__ AL,
                  unsigned short* __restrict__ BH, unsigned short* __restrict__ BL) {
  const int row  = blockIdx.x;              // 0 .. 2*MROWS-1 (A rows then B rows)
  const int lane = threadIdx.x;             // 0..31
  const float* src;
  unsigned short *dH, *dL;
  int i;
  if (row < MROWS) { i = row;         src = A; dH = AH; dL = AL; }
  else             { i = row - MROWS; src = B; dH = BH; dL = BL; }
  const int n = i & (NSMP - 1);             // sample index
  const int v = i >> 12;                    // view index (NSMP = 2^12)
  const float* p = src + ((size_t)(n * NVIEW + v)) * DDIM + lane * 8;

  float x[8];
  float ss = 0.0f;
#pragma unroll
  for (int e = 0; e < 8; ++e) { x[e] = p[e]; ss += x[e] * x[e]; }
#pragma unroll
  for (int m = 1; m <= 16; m <<= 1) ss += __shfl_xor(ss, m, 32);
  const float inv = 1.0f / fmaxf(sqrtf(ss), 1e-12f);

  ushort8 ho, lo;
#pragma unroll
  for (int e = 0; e < 8; ++e) {
    const float xv = x[e] * inv;
    const unsigned short h = f2bf(xv);
    ho[e] = h;
    lo[e] = f2bf(xv - bf2f(h));
  }
  const size_t off = (size_t)i * DDIM + lane * 8;   // 16B aligned
  *reinterpret_cast<ushort8*>(dH + off) = ho;
  *reinterpret_cast<ushort8*>(dL + off) = lo;
}

// ---------------------------------------------------------------------------
// Pass 2: 8 waves/block, each wave owns one 16-row A tile (held in VGPRs);
// B tiles (hi+lo) are double-buffered through LDS: loads for tile ct+1 are
// issued into VGPRs before the WMMAs of tile ct (latency hidden by matrix
// work), ds_store lands in the other buffer, single barrier per tile.
// ---------------------------------------------------------------------------
__global__ void __launch_bounds__(256)
contrastive_main(const unsigned short* __restrict__ AH, const unsigned short* __restrict__ AL,
                 const unsigned short* __restrict__ BH, const unsigned short* __restrict__ BL,
                 float* __restrict__ psum, float* __restrict__ pdiag,
                 float* __restrict__ ppos) {
  __shared__ unsigned short lB[2][2][16 * LDSPITCH];   // [buf][H/L][row][256+pad]

  const int tid  = threadIdx.x;
  const int wv   = tid >> 5;                 // wave 0..7
  const int lane = tid & 31;
  const int lm   = lane & 15;
  const int hi   = lane >> 4;
  const int rt   = blockIdx.x * 8 + wv;      // row tile 0..511
  const int chunk = blockIdx.y;              // 0..NCHUNK-1
  const int ct0  = chunk * CTPER;
  const int ctEnd = ct0 + CTPER;
  const float shift = INV_T;                 // |cos| <= 1 -> s/T - 1/T <= ~0
  const float LOG2E = 1.44269504088896f;

  // Staging coordinates for this thread (2 chunks of 8 ushorts per plane).
  int srow[2], soff[2];
#pragma unroll
  for (int j = 0; j < 2; ++j) {
    const int q = tid + j * 256;             // 0..511
    srow[j] = q >> 5;                        // 0..15
    soff[j] = (q & 31) * 8;                  // ushort offset 0..248
  }

  // Preload this wave's A operands for all 8 K-steps (WMMA A layout §7.12.2:
  // row = lane%16; K runs {8*hi..8*hi+7} and {16+8*hi..16+8*hi+7} per step).
  v16bf aH[8], aL[8];
  {
    const size_t arow = (size_t)(rt * 16 + lm) * DDIM;
#pragma unroll
    for (int k = 0; k < 8; ++k) {
      const size_t k0 = arow + k * 32 + 8 * hi;
      union { ushort8 h[2]; v16bf v; } ph, pl;
      ph.h[0] = *reinterpret_cast<const ushort8*>(AH + k0);
      ph.h[1] = *reinterpret_cast<const ushort8*>(AH + k0 + 16);
      pl.h[0] = *reinterpret_cast<const ushort8*>(AL + k0);
      pl.h[1] = *reinterpret_cast<const ushort8*>(AL + k0 + 16);
      aH[k] = ph.v;
      aL[k] = pl.v;
    }
  }

  float sum[8], diagp[8], posv[8];
#pragma unroll
  for (int r = 0; r < 8; ++r) { sum[r] = 0.0f; diagp[r] = 0.0f; posv[r] = 0.0f; }

  const int diagT = rt;                                  // tile holding s_ii
  const int posT  = (rt + NTILES / 2) & (NTILES - 1);    // tile holding positives

  // Prologue: stage first tile into buffer 0.
  ushort8 rh[2], rl[2];
#pragma unroll
  for (int j = 0; j < 2; ++j) {
    const size_t g = (size_t)(ct0 * 16 + srow[j]) * DDIM + soff[j];
    rh[j] = *reinterpret_cast<const ushort8*>(BH + g);
    rl[j] = *reinterpret_cast<const ushort8*>(BL + g);
  }
#pragma unroll
  for (int j = 0; j < 2; ++j) {
    *reinterpret_cast<ushort8*>(&lB[0][0][srow[j] * LDSPITCH + soff[j]]) = rh[j];
    *reinterpret_cast<ushort8*>(&lB[0][1][srow[j] * LDSPITCH + soff[j]]) = rl[j];
  }
  __syncthreads();

  for (int ct = ct0; ct < ctEnd; ++ct) {
    const int buf = (ct - ct0) & 1;
    const bool hasNext = (ct + 1 < ctEnd);

    // Issue global loads for the NEXT tile now; WMMAs below hide the latency.
    if (hasNext) {
#pragma unroll
      for (int j = 0; j < 2; ++j) {
        const size_t g = (size_t)((ct + 1) * 16 + srow[j]) * DDIM + soff[j];
        rh[j] = *reinterpret_cast<const ushort8*>(BH + g);
        rl[j] = *reinterpret_cast<const ushort8*>(BL + g);
      }
    }

    // ---- 16x16 logits tile: s = aH*bH + aH*bL + aL*bH over K=256 ----
    const unsigned short* lbh = &lB[buf][0][lm * LDSPITCH];
    const unsigned short* lbl = &lB[buf][1][lm * LDSPITCH];
    v8f C = {};
#pragma unroll
    for (int k = 0; k < 8; ++k) {
      const int k0 = k * 32 + 8 * hi;       // B layout: K {8*hi..8*hi+7} and {16+8*hi..16+8*hi+7}, col = lane%16
      union { ushort8 h[2]; v16bf v; } bh, bl;
      bh.h[0] = *reinterpret_cast<const ushort8*>(lbh + k0);
      bh.h[1] = *reinterpret_cast<const ushort8*>(lbh + k0 + 16);
      bl.h[0] = *reinterpret_cast<const ushort8*>(lbl + k0);
      bl.h[1] = *reinterpret_cast<const ushort8*>(lbl + k0 + 16);
      C = __builtin_amdgcn_wmma_f32_16x16x32_bf16(false, aH[k], false, bh.v,
                                                  (short)0, C, false, false);
      C = __builtin_amdgcn_wmma_f32_16x16x32_bf16(false, aH[k], false, bl.v,
                                                  (short)0, C, false, false);
      C = __builtin_amdgcn_wmma_f32_16x16x32_bf16(false, aL[k], false, bh.v,
                                                  (short)0, C, false, false);
    }

    // C/D layout: VGPR r holds row m = r + 8*hi, column = lane%16.
#pragma unroll
    for (int r = 0; r < 8; ++r) {
      const float s = C[r];
      const float p = exp2f((s * INV_T - shift) * LOG2E);
      sum[r] += p;
      const int m = r + 8 * hi;
      if (ct == diagT && lm == m) diagp[r] = p;   // exclude diagonal later
      if (ct == posT  && lm == m) posv[r]  = s;   // positive logit
    }

    // Store next tile into the other buffer. The barrier below both publishes
    // these stores and guarantees buf^1's previous readers are finished.
    if (hasNext) {
#pragma unroll
      for (int j = 0; j < 2; ++j) {
        *reinterpret_cast<ushort8*>(&lB[buf ^ 1][0][srow[j] * LDSPITCH + soff[j]]) = rh[j];
        *reinterpret_cast<ushort8*>(&lB[buf ^ 1][1][srow[j] * LDSPITCH + soff[j]]) = rl[j];
      }
    }
    __syncthreads();
  }

  // Cross-lane totals within each 16-lane half (wave32: masks 1..8 stay inside).
#pragma unroll
  for (int r = 0; r < 8; ++r) {
#pragma unroll
    for (int m = 1; m <= 8; m <<= 1) {
      sum[r]   += __shfl_xor(sum[r],   m, 32);
      diagp[r] += __shfl_xor(diagp[r], m, 32);
      posv[r]  += __shfl_xor(posv[r],  m, 32);
    }
  }
  if (lm == 0) {                 // lane 0 -> rows 0..7, lane 16 -> rows 8..15
#pragma unroll
    for (int r = 0; r < 8; ++r) {
      const int m   = r + 8 * hi;
      const int row = rt * 16 + m;
      psum [chunk * MROWS + row] = sum[r];
      pdiag[chunk * MROWS + row] = diagp[r];
      ppos [chunk * MROWS + row] = posv[r];
    }
  }
}

// ---------------------------------------------------------------------------
// Pass 3: combine chunks per row, per-row loss, deterministic mean -> scalar.
// ---------------------------------------------------------------------------
__global__ void __launch_bounds__(256)
finalize_kernel(const float* __restrict__ psum, const float* __restrict__ pdiag,
                const float* __restrict__ ppos, float* __restrict__ out) {
  __shared__ float sm[256];
  const int t = threadIdx.x;
  float acc = 0.0f;
  for (int i = t; i < MROWS; i += 256) {
    float s = 0.0f, d = 0.0f, p = 0.0f;
#pragma unroll
    for (int c = 0; c < NCHUNK; ++c) {
      s += psum [c * MROWS + i];
      d += pdiag[c * MROWS + i];
      p += ppos [c * MROWS + i];
    }
    // logsumexp_{j != i} = log(s - d) + 1/T ; loss = lse - pos/T
    acc += logf(s - d) + INV_T - p * INV_T;
  }
  sm[t] = acc;
  __syncthreads();
  for (int w = 128; w > 0; w >>= 1) {
    if (t < w) sm[t] += sm[t + w];
    __syncthreads();
  }
  if (t == 0) out[0] = sm[0] / (float)MROWS;
}

extern "C" void kernel_launch(void* const* d_in, const int* in_sizes, int n_in,
                              void* d_out, int out_size, void* d_ws, size_t ws_size,
                              hipStream_t stream) {
  (void)in_sizes; (void)n_in; (void)out_size; (void)ws_size;
  const float* A = (const float*)d_in[0];
  const float* B = (const float*)d_in[1];

  char* ws = (char*)d_ws;
  const size_t plane = (size_t)MROWS * DDIM * sizeof(unsigned short);  // 4 MB
  unsigned short* AH = (unsigned short*)(ws);
  unsigned short* AL = (unsigned short*)(ws + plane);
  unsigned short* BH = (unsigned short*)(ws + 2 * plane);
  unsigned short* BL = (unsigned short*)(ws + 3 * plane);
  float* psum  = (float*)(ws + 4 * plane);                              // 128 KB
  float* pdiag = (float*)(ws + 4 * plane + (size_t)NCHUNK * MROWS * 4);
  float* ppos  = (float*)(ws + 4 * plane + (size_t)2 * NCHUNK * MROWS * 4);

  norm_split_kernel<<<2 * MROWS, 32, 0, stream>>>(A, B, AH, AL, BH, BL);
  contrastive_main<<<dim3(NTILES / 8, NCHUNK), 256, 0, stream>>>(
      AH, AL, BH, BL, psum, pdiag, ppos);
  finalize_kernel<<<1, 256, 0, stream>>>(psum, pdiag, ppos, (float*)d_out);
}
